// MultiHeadAttention_59193239273723
// MI455X (gfx1250) — matrix operand order repaired and checked by reading, not compiled, not run on hardware
//
#include <hip/hip_runtime.h>
#include <hip/hip_bf16.h>

typedef __attribute__((ext_vector_type(16))) __bf16 v16bf;
typedef __attribute__((ext_vector_type(8)))  float  v8f;
typedef int v4i __attribute__((vector_size(4 * sizeof(int))));

#define D_MODEL 768
#define SEQ_N   2048
#define N_HEADS 12
#define D_K     64

#define AS1 __attribute__((address_space(1)))
#define AS3 __attribute__((address_space(3)))

#if __has_builtin(__builtin_amdgcn_global_load_async_to_lds_b128) && \
    __has_builtin(__builtin_amdgcn_s_wait_asynccnt)
#define HAS_ASYNC_LDS 1
#else
#define HAS_ASYNC_LDS 0
#endif

union BF16Frag { v16bf v; unsigned u[8]; };

__device__ __forceinline__ unsigned short f2bf(float x) {
    union { float f; unsigned u; } c; c.f = x;
    unsigned r = c.u + 0x7FFFu + ((c.u >> 16) & 1u);   // round-to-nearest-even
    return (unsigned short)(r >> 16);
}

// ---- WMMA fragment loaders from LDS (bf16, stride in dwords) -------------
// A 16x32 (MxK): lane m = lane&15 (+row base); lanes>=16 take K offset +8.
__device__ __forceinline__ v16bf load_frag_a(const unsigned short* lds, int row_base,
                                             int stride_dw, int k_base, int lane) {
    const unsigned* d = (const unsigned*)lds;
    int m    = row_base + (lane & 15);
    int koff = (lane >> 4) * 8;
    BF16Frag f;
    int b0 = m * stride_dw + ((k_base + koff) >> 1);
#pragma unroll
    for (int i = 0; i < 4; ++i) f.u[i] = d[b0 + i];
    int b1 = m * stride_dw + ((k_base + koff + 16) >> 1);
#pragma unroll
    for (int i = 0; i < 4; ++i) f.u[4 + i] = d[b1 + i];
    return f.v;
}

// B 32x16 (KxN) from LDS stored transposed as [n][k]: lane n = lane&15 (+base);
// lanes>=16 take K offset +8.
__device__ __forceinline__ v16bf load_frag_b(const unsigned short* lds, int n_base,
                                             int stride_dw, int k_base, int lane) {
    const unsigned* d = (const unsigned*)lds;
    int n    = n_base + (lane & 15);
    int koff = (lane >> 4) * 8;
    BF16Frag f;
    int b0 = n * stride_dw + ((k_base + koff) >> 1);
#pragma unroll
    for (int i = 0; i < 4; ++i) f.u[i] = d[b0 + i];
    int b1 = n * stride_dw + ((k_base + koff + 16) >> 1);
#pragma unroll
    for (int i = 0; i < 4; ++i) f.u[4 + i] = d[b1 + i];
    return f.v;
}

// ==========================================================================
// GEMM: C[8192 x 768] = A * W + bias.  Block 128 thr (4 waves), 64x64 tile,
// BK=64 (2 WMMA k-steps per staging round -> 8 WMMAs per barrier pair).
// OUT_MODE 0: f32 row-major | 1: bf16 [B,H,N,64] | 2: bf16 [B,H,64,N]
// ==========================================================================
template <int OUT_MODE, bool A_BF16>
__global__ __launch_bounds__(128) void gemm768(const void* __restrict__ Aptr,
                                               const float* __restrict__ W,
                                               const float* __restrict__ bias,
                                               void* __restrict__ Out) {
    __shared__ alignas(16) unsigned short a_lds[64 * 72];   // [m][k], 36-dword stride
    __shared__ alignas(16) unsigned short b_lds[64 * 72];   // W transposed [n][k]

    const int tid  = threadIdx.x;
    const int lane = tid & 31;
    const int wave = tid >> 5;
    const int n0   = blockIdx.x * 64;
    const int row0 = blockIdx.y * 64;

    v8f acc[4] = {};

    for (int k0 = 0; k0 < D_MODEL; k0 += 64) {
        if (k0 + 64 < D_MODEL)   // gfx1250 global_prefetch of next A tile
            __builtin_prefetch((const char*)Aptr +
                ((size_t)(row0 + (tid >> 1)) * D_MODEL + k0 + 64) * (A_BF16 ? 2 : 4), 0, 0);
        // stage A tile 64x64 -> bf16
#pragma unroll
        for (int i = 0; i < 32; ++i) {
            int e = tid + i * 128;
            int r = e >> 6, kk = e & 63;
            unsigned short hv;
            if constexpr (A_BF16)
                hv = ((const unsigned short*)Aptr)[(size_t)(row0 + r) * D_MODEL + k0 + kk];
            else
                hv = f2bf(((const float*)Aptr)[(size_t)(row0 + r) * D_MODEL + k0 + kk]);
            a_lds[r * 72 + kk] = hv;
        }
        // stage W tile 64x64, transposed into [n][k]
#pragma unroll
        for (int i = 0; i < 32; ++i) {
            int e = tid + i * 128;
            int kk = e >> 6, n = e & 63;
            b_lds[n * 72 + kk] = f2bf(W[(size_t)(k0 + kk) * D_MODEL + n0 + n]);
        }
        __syncthreads();
#pragma unroll
        for (int ks = 0; ks < 64; ks += 32) {
            v16bf af = load_frag_a(a_lds, wave * 16, 36, ks, lane);
#pragma unroll
            for (int ns = 0; ns < 4; ++ns) {
                v16bf bfr = load_frag_b(b_lds, ns * 16, 36, ks, lane);
                acc[ns] = __builtin_amdgcn_wmma_f32_16x16x32_bf16(
                    false, af, false, bfr, (short)0, acc[ns], false, false);
            }
        }
        __syncthreads();
    }

    // epilogue: C layout -> lane n = lane&15, VGPR r -> M = r + 8*(lane>=16)
    const int half = lane >> 4;
    const int nn   = lane & 15;
#pragma unroll
    for (int ns = 0; ns < 4; ++ns) {
#pragma unroll
        for (int r = 0; r < 8; ++r) {
            int m   = wave * 16 + r + half * 8;
            int row = row0 + m;
            int col = n0 + ns * 16 + nn;
            float v = acc[ns][r] + bias[col];
            if constexpr (OUT_MODE == 0) {
                ((float*)Out)[(size_t)row * D_MODEL + col] = v;
            } else {
                int b  = row >> 11, nq = row & (SEQ_N - 1);
                int h  = col >> 6,  dk = col & 63;
                if constexpr (OUT_MODE == 1)
                    ((unsigned short*)Out)[((((size_t)b * N_HEADS + h) * SEQ_N + nq) << 6) + dk] = f2bf(v);
                else
                    ((unsigned short*)Out)[(((size_t)b * N_HEADS + h) * D_K + dk) * SEQ_N + nq] = f2bf(v);
            }
        }
    }
}

// ---- K/V tile staging: async DMA to LDS (ASYNCcnt) with sync fallback ----
__device__ __forceinline__ void stage_tiles(const unsigned short* __restrict__ Kh,
                                            const unsigned short* __restrict__ Vt,
                                            size_t headQK, size_t headV, int kt,
                                            unsigned short* kbuf, unsigned short* vbuf,
                                            int tid) {
#if HAS_ASYNC_LDS
#pragma unroll
    for (int i = 0; i < 4; ++i) {
        int c   = tid + i * 128;           // 512 x b128 per tile
        int row = c >> 3, seg = c & 7;     // row 0..63, seg*8 bf16 = 16B
        __builtin_amdgcn_global_load_async_to_lds_b128(
            (AS1 v4i*)(void*)(Kh + headQK + (size_t)(kt + row) * D_K + seg * 8),
            (AS3 v4i*)(void*)(kbuf + row * 72 + seg * 8), 0, 0);
        __builtin_amdgcn_global_load_async_to_lds_b128(
            (AS1 v4i*)(void*)(Vt + headV + (size_t)row * SEQ_N + kt + seg * 8),
            (AS3 v4i*)(void*)(vbuf + row * 72 + seg * 8), 0, 0);
    }
#else
#pragma unroll
    for (int i = 0; i < 32; ++i) {
        int e = tid + i * 128;
        int r = e >> 6, c = e & 63;
        kbuf[r * 72 + c] = Kh[headQK + (size_t)(kt + r) * D_K + c];
        vbuf[r * 72 + c] = Vt[headV + (size_t)r * SEQ_N + kt + c];
    }
#endif
}

__device__ __forceinline__ void wait_tiles() {
#if HAS_ASYNC_LDS
    __builtin_amdgcn_s_wait_asynccnt(0);
#endif
}

// ==========================================================================
// Flash attention: per block (b, h, 64 q-rows); double-buffered 64-key tiles
// streamed via async-to-LDS DMA.  Qh,Kh: bf16 [B,H,N,64]; Vt: bf16 [B,H,64,N]
// ==========================================================================
__global__ __launch_bounds__(128) void attn_kernel(const unsigned short* __restrict__ Qh,
                                                   const unsigned short* __restrict__ Kh,
                                                   const unsigned short* __restrict__ Vt,
                                                   unsigned short* __restrict__ Ao) {
    __shared__ alignas(16) unsigned short q_lds[64 * 72];      // [m][dk]
    __shared__ alignas(16) unsigned short k_buf[2][64 * 72];   // [key][dk] (B-frag)
    __shared__ alignas(16) unsigned short v_buf[2][64 * 72];   // [dk][key] (B-frag)
    __shared__ alignas(16) unsigned short p_lds[4][16 * 72];   // per-wave P buffer

    const int tid  = threadIdx.x;
    const int lane = tid & 31;
    const int wave = tid >> 5;
    const int half = lane >> 4;
    const int nn   = lane & 15;

    const int q0 = blockIdx.x * 64;
    const int h  = blockIdx.y;
    const int b  = blockIdx.z;

    const size_t headQK = (((size_t)b * N_HEADS + h) * SEQ_N) * D_K;
    const size_t headV  = (((size_t)b * N_HEADS + h) * D_K) * SEQ_N;

    // kick off DMA for key-tile 0 while we stage Q through VGPRs
    stage_tiles(Kh, Vt, headQK, headV, 0, k_buf[0], v_buf[0], tid);

#pragma unroll
    for (int i = 0; i < 32; ++i) {
        int e = tid + i * 128;
        int r = e >> 6, dk = e & 63;
        q_lds[r * 72 + dk] = Qh[headQK + (size_t)(q0 + r) * D_K + dk];
    }

    v8f o[4] = {};
    float mrow[8], lrow[8];
#pragma unroll
    for (int r = 0; r < 8; ++r) { mrow[r] = -1e30f; lrow[r] = 0.0f; }

    for (int kt = 0; kt < SEQ_N; kt += 64) {
        const int cur = (kt >> 6) & 1;
        wait_tiles();          // our async loads for buf[cur] (and older) complete
        __syncthreads();       // all waves' loads visible; prev compute on buf[1-cur] done
        if (kt + 64 < SEQ_N)   // prefetch next tile into the other buffer
            stage_tiles(Kh, Vt, headQK, headV, kt + 64, k_buf[1 - cur], v_buf[1 - cur], tid);

        // S = Q * K^T  (16 q-rows per wave x 64 keys)
        v8f s[4] = {};
#pragma unroll
        for (int kk = 0; kk < D_K; kk += 32) {
            v16bf qf = load_frag_a(q_lds, wave * 16, 36, kk, lane);
#pragma unroll
            for (int ns = 0; ns < 4; ++ns) {
                v16bf kf = load_frag_b(k_buf[cur], ns * 16, 36, kk, lane);
                s[ns] = __builtin_amdgcn_wmma_f32_16x16x32_bf16(
                    false, qf, false, kf, (short)0, s[ns], false, false);
            }
        }

        // online softmax; row m = r + 8*half lives across 16 lanes (nn) x 4 accs
#pragma unroll
        for (int r = 0; r < 8; ++r) {
            float mx = -1e30f;
#pragma unroll
            for (int ns = 0; ns < 4; ++ns) mx = fmaxf(mx, s[ns][r] * 0.125f);
#pragma unroll
            for (int off = 8; off >= 1; off >>= 1) mx = fmaxf(mx, __shfl_xor(mx, off, 16));
            float mnew  = fmaxf(mrow[r], mx);
            float alpha = __expf(mrow[r] - mnew);
            float psum  = 0.0f;
#pragma unroll
            for (int ns = 0; ns < 4; ++ns) {
                float p = __expf(s[ns][r] * 0.125f - mnew);
                s[ns][r] = p;
                psum += p;
            }
#pragma unroll
            for (int off = 8; off >= 1; off >>= 1) psum += __shfl_xor(psum, off, 16);
            lrow[r] = lrow[r] * alpha + psum;
            mrow[r] = mnew;
#pragma unroll
            for (int ns = 0; ns < 4; ++ns) o[ns][r] *= alpha;
            // C-frag -> A-frag re-layout through per-wave LDS (same-wave order)
#pragma unroll
            for (int ns = 0; ns < 4; ++ns)
                p_lds[wave][(r + 8 * half) * 72 + ns * 16 + nn] = f2bf(s[ns][r]);
        }

        // O += P * V
#pragma unroll
        for (int kk = 0; kk < 64; kk += 32) {
            v16bf pf = load_frag_a(p_lds[wave], 0, 36, kk, lane);
#pragma unroll
            for (int ns = 0; ns < 4; ++ns) {
                v16bf vf = load_frag_b(v_buf[cur], ns * 16, 36, kk, lane);
                o[ns] = __builtin_amdgcn_wmma_f32_16x16x32_bf16(
                    false, pf, false, vf, (short)0, o[ns], false, false);
            }
        }
    }

    // epilogue: normalize by l, write bf16 Ao[b*N + q, h*64 + dk]
#pragma unroll
    for (int r = 0; r < 8; ++r) {
        float inv = 1.0f / lrow[r];
        int m = wave * 16 + r + 8 * half;
#pragma unroll
        for (int ns = 0; ns < 4; ++ns) {
            int dk = ns * 16 + nn;
            Ao[((size_t)b * SEQ_N + q0 + m) * D_MODEL + h * D_K + dk] = f2bf(o[ns][r] * inv);
        }
    }
}

// ==========================================================================
extern "C" void kernel_launch(void* const* d_in, const int* in_sizes, int n_in,
                              void* d_out, int out_size, void* d_ws, size_t ws_size,
                              hipStream_t stream) {
    const float* q  = (const float*)d_in[0];
    const float* k  = (const float*)d_in[1];
    const float* v  = (const float*)d_in[2];
    const float* Wq = (const float*)d_in[3];
    const float* bq = (const float*)d_in[4];
    const float* Wk = (const float*)d_in[5];
    const float* bk = (const float*)d_in[6];
    const float* Wv = (const float*)d_in[7];
    const float* bv = (const float*)d_in[8];
    const float* Wo = (const float*)d_in[9];
    const float* bo = (const float*)d_in[10];

    const size_t elems = (size_t)4 * N_HEADS * SEQ_N * D_K;   // 6,291,456 bf16 each
    unsigned short* Qh = (unsigned short*)d_ws;
    unsigned short* Kh = Qh + elems;
    unsigned short* Vt = Kh + elems;
    unsigned short* Ao = Vt + elems;

    dim3 gg(D_MODEL / 64, (4 * SEQ_N) / 64);   // 12 x 128 blocks

    gemm768<1, false><<<gg, 128, 0, stream>>>(q, Wq, bq, Qh);
    gemm768<1, false><<<gg, 128, 0, stream>>>(k, Wk, bk, Kh);
    gemm768<2, false><<<gg, 128, 0, stream>>>(v, Wv, bv, Vt);

    attn_kernel<<<dim3(SEQ_N / 64, N_HEADS, 4), 128, 0, stream>>>(Qh, Kh, Vt, Ao);

    gemm768<0, true><<<gg, 128, 0, stream>>>(Ao, Wo, bo, d_out);
}
